// DeepFM_66331474919973
// MI455X (gfx1250) — matrix operand order repaired and checked by reading, not compiled, not run on hardware
//
#include <hip/hip_runtime.h>
#include <stdint.h>

// ---------------------------------------------------------------------------
// DeepFM inference for MI455X (gfx1250, wave32, WMMA, fused MLP,
// double-buffered fragment loads).
// ---------------------------------------------------------------------------

typedef __attribute__((ext_vector_type(16))) __bf16 v16bf;
typedef __attribute__((ext_vector_type(8)))  __bf16 v8bf;
typedef __attribute__((ext_vector_type(8)))  float  v8f;

#define B_ROWS 16384
#define KDIM   64
#define LC     8
#define LT     64
#define FIELDS 6
#define DX     (FIELDS * KDIM)   // 384
#define H1     128
#define H2     64

// padded LDS row strides (bf16 elements) to rotate banks between rows
#define XS_LD  (DX + 8)          // 392 -> 784B row stride, 16B aligned
#define H1_LD  (H1 + 8)          // 136 -> 272B row stride
#define H2_LD  (H2 + 8)          // 72

// float -> bf16 bits, round to nearest even
__device__ __forceinline__ uint16_t f2bf(float f) {
    union { float f; uint32_t u; } v; v.f = f;
    uint32_t r = v.u + 0x7FFFu + ((v.u >> 16) & 1u);
    return (uint16_t)(r >> 16);
}
__device__ __forceinline__ float bf2f(uint16_t b) {
    union { uint32_t u; float f; } v; v.u = ((uint32_t)b) << 16;
    return v.f;
}
__device__ __forceinline__ v16bf combine16(v8bf lo, v8bf hi) {
    v16bf r;
#pragma unroll
    for (int i = 0; i < 8; ++i) { r[i] = lo[i]; r[8 + i] = hi[i]; }
    return r;
}
// A fragment (16x32 bf16): base already includes row*ld + (half?16:0)
__device__ __forceinline__ v16bf loadA(const __bf16* base, int k0) {
    return combine16(*(const v8bf*)(base + k0), *(const v8bf*)(base + k0 + 8));
}
// B fragment (32x16 bf16 from W^T): base already includes col*ld + (half?16:0)
__device__ __forceinline__ v16bf loadB(const __bf16* base, int k0) {
    return combine16(*(const v8bf*)(base + k0), *(const v8bf*)(base + k0 + 8));
}

// ---------------------------------------------------------------------------
// Convert W1 [384,128] and W2 [128,64] (fp32, row-major K x N) into
// transposed bf16 [N][K] so WMMA B fragments are contiguous-K loads.
// ---------------------------------------------------------------------------
__global__ void convert_w(const float* __restrict__ W1, const float* __restrict__ W2,
                          uint16_t* __restrict__ W1t, uint16_t* __restrict__ W2t) {
    int i = blockIdx.x * blockDim.x + threadIdx.x;
    if (i < DX * H1) {
        int k = i / H1, n = i % H1;
        W1t[n * DX + k] = f2bf(W1[i]);
    } else if (i < DX * H1 + H1 * H2) {
        int j = i - DX * H1;
        int k = j / H2, n = j % H2;
        W2t[n * H1 + k] = f2bf(W2[j]);
    }
}

// ---------------------------------------------------------------------------
// Per-row embedding gather + masked mean pooling + first-order + FM2.
// One block of 64 threads per row; thread k owns embedding dim k.
// Writes bf16 deep_x [B,384] and float partial logits [B].
// ---------------------------------------------------------------------------
__global__ void pool_kernel(
    const int* __restrict__ user, const int* __restrict__ item, const int* __restrict__ brand,
    const int* __restrict__ cat_idx, const uint8_t* __restrict__ cat_mask,
    const int* __restrict__ text_idx, const uint8_t* __restrict__ text_mask,
    const float* __restrict__ sales_rank,
    const float* __restrict__ user_w, const float* __restrict__ item_w,
    const float* __restrict__ brand_w, const float* __restrict__ cat_w,
    const float* __restrict__ text_w,
    const float* __restrict__ sproj_w, const float* __restrict__ sproj_b,
    const float* __restrict__ user_l, const float* __restrict__ item_l,
    const float* __restrict__ brand_l, const float* __restrict__ cat_l,
    const float* __restrict__ text_l,
    const float* __restrict__ sales_lw, const float* __restrict__ sales_lb,
    uint16_t* __restrict__ deep_x, float* __restrict__ logit_part)
{
    const int b = blockIdx.x;
    const int k = threadIdx.x;           // 0..63
    const int u  = user[b];
    const int it = item[b];
    const int br = brand[b];

    float vu = user_w [(size_t)u  * KDIM + k];
    float vi = item_w [(size_t)it * KDIM + k];
    float vb = brand_w[(size_t)br * KDIM + k];

    // masked mean over categories (mask is wave-uniform -> branch skips loads)
    float vc = 0.f, ccnt = 0.f;
#pragma unroll
    for (int l = 0; l < LC; ++l) {
        int m = cat_mask[b * LC + l];
        if (m) vc += cat_w[(size_t)cat_idx[b * LC + l] * KDIM + k];
        ccnt += (float)m;
    }
    vc /= fmaxf(ccnt, 1.0f);

    // masked mean over text tokens (dominant HBM traffic, coalesced across k)
    float vt = 0.f, tcnt = 0.f;
    for (int l = 0; l < LT; ++l) {
        int m = text_mask[b * LT + l];
        if (m) vt += text_w[(size_t)text_idx[b * LT + l] * KDIM + k];
        tcnt += (float)m;
    }
    vt /= fmaxf(tcnt, 1.0f);

    const float sr = sales_rank[b];
    const float vs = sr * sproj_w[k] + sproj_b[k];

    // bf16 deep_x, field order matches jnp.stack([...], 1).reshape
    uint16_t* dx = deep_x + (size_t)b * DX;
    dx[0 * KDIM + k] = f2bf(vu);
    dx[1 * KDIM + k] = f2bf(vi);
    dx[2 * KDIM + k] = f2bf(vb);
    dx[3 * KDIM + k] = f2bf(vc);
    dx[4 * KDIM + k] = f2bf(vt);
    dx[5 * KDIM + k] = f2bf(vs);

    // FM second-order partials (fp32, pre-conversion values)
    const float sv = vu + vi + vb + vc + vt + vs;
    const float sq = vu*vu + vi*vi + vb*vb + vc*vc + vt*vt + vs*vs;

    // distributed first-order linear partials: thread k handles text token k,
    // threads 0..7 also handle category slot k
    float lin = text_mask[b * LT + k] ? text_l[text_idx[b * LT + k]] : 0.f;
    if (k < LC) lin += cat_mask[b * LC + k] ? cat_l[cat_idx[b * LC + k]] : 0.f;

    __shared__ float r1[64], r2[64], r3[64];
    r1[k] = sv * sv; r2[k] = sq; r3[k] = lin;
    __syncthreads();
#pragma unroll
    for (int s = 32; s > 0; s >>= 1) {
        if (k < s) { r1[k] += r1[k + s]; r2[k] += r2[k + s]; r3[k] += r3[k + s]; }
        __syncthreads();
    }
    if (k == 0) {
        float first = user_l[u] + item_l[it] + brand_l[br] + r3[0]
                    + sr * sales_lw[0] + sales_lb[0];
        float fm2 = 0.5f * (r1[0] - r2[0]);
        logit_part[b] = first + fm2;
    }
}

// ---------------------------------------------------------------------------
// Fully fused MLP: deep_x [B,384] -> relu(x@W1+b1) -> relu(h@W2+b2) -> @W3+b3
// -> + logit_part -> sigmoid -> out.
// One block (256 threads = 8 wave32) per 16-row tile; grid = B/16 = 1024.
// K-loops are double-buffered: iteration i+1's A/B fragments are issued
// before iteration i's WMMA so loads overlap matrix ops instead of the
// fully-serialized s_wait_loadcnt_dscnt 0x0 pattern.
// ---------------------------------------------------------------------------
__global__ void fused_mlp(const uint16_t* __restrict__ deep_x,
                          const uint16_t* __restrict__ W1t,   // [128][384] bf16
                          const uint16_t* __restrict__ W2t,   // [64][128] bf16
                          const float* __restrict__ b1,
                          const float* __restrict__ b2,
                          const float* __restrict__ W3,
                          const float* __restrict__ b3,
                          const float* __restrict__ logit_part,
                          float* __restrict__ out)
{
    __shared__ uint16_t xs [16 * XS_LD];   // 12.25 KB
    __shared__ uint16_t h1s[16 * H1_LD];   //  4.25 KB
    __shared__ uint16_t h2s[16 * H2_LD];   //  2.25 KB

    const int tid  = threadIdx.x;
    const int lane = tid & 31;
    const int wave = tid >> 5;             // 0..7
    const int tm   = blockIdx.x * 16;
    const int half = lane >> 4;            // 0: lanes 0..15, 1: lanes 16..31
    const int l16  = lane & 15;

    // ---- stage deep_x tile into LDS (16 rows x 384 bf16 = 48 x16B per row) -
    {
        const uint4* src = (const uint4*)(deep_x + (size_t)tm * DX);
#pragma unroll
        for (int c = tid; c < 16 * (DX / 8); c += 256) {     // 768 chunks
            int row = c / (DX / 8), col = c % (DX / 8);
            *(uint4*)&xs[row * XS_LD + col * 8] = src[row * (DX / 8) + col];
        }
    }
    __syncthreads();

    // ---- layer 1: each wave computes cols [wave*16, wave*16+16) ------------
    {
        const int tn = wave * 16;
        const __bf16* Ap = (const __bf16*)&xs[l16 * XS_LD + (half ? 16 : 0)];
        const __bf16* Bp = (const __bf16*)W1t + (size_t)(tn + l16) * DX + (half ? 16 : 0);
        const float bia = b1[tn + l16];

        v8f acc = {};
        v16bf av = loadA(Ap, 0);
        v16bf bv = loadB(Bp, 0);
#pragma unroll
        for (int k0 = 0; k0 < DX; k0 += 32) {
            v16bf an = av, bn = bv;
            if (k0 + 32 < DX) {           // compile-time (fully unrolled)
                an = loadA(Ap, k0 + 32);
                bn = loadB(Bp, k0 + 32);
            }
            acc = __builtin_amdgcn_wmma_f32_16x16x32_bf16(
                      false, av, false, bv, (short)0, acc, false, false);
            av = an; bv = bn;
        }
        const int rbase = half ? 8 : 0;
#pragma unroll
        for (int r = 0; r < 8; ++r)
            h1s[(rbase + r) * H1_LD + tn + l16] = f2bf(fmaxf(acc[r] + bia, 0.0f));
    }
    __syncthreads();

    // ---- layer 2: waves 0..3 compute cols [wave*16, wave*16+16) of 64 ------
    if (wave < 4) {
        const int tn = wave * 16;
        const __bf16* Ap = (const __bf16*)&h1s[l16 * H1_LD + (half ? 16 : 0)];
        const __bf16* Bp = (const __bf16*)W2t + (size_t)(tn + l16) * H1 + (half ? 16 : 0);
        const float bia = b2[tn + l16];

        v8f acc = {};
        v16bf av = loadA(Ap, 0);
        v16bf bv = loadB(Bp, 0);
#pragma unroll
        for (int k0 = 0; k0 < H1; k0 += 32) {
            v16bf an = av, bn = bv;
            if (k0 + 32 < H1) {
                an = loadA(Ap, k0 + 32);
                bn = loadB(Bp, k0 + 32);
            }
            acc = __builtin_amdgcn_wmma_f32_16x16x32_bf16(
                      false, av, false, bv, (short)0, acc, false, false);
            av = an; bv = bn;
        }
        const int rbase = half ? 8 : 0;
#pragma unroll
        for (int r = 0; r < 8; ++r)
            h2s[(rbase + r) * H2_LD + tn + l16] = f2bf(fmaxf(acc[r] + bia, 0.0f));
    }
    __syncthreads();

    // ---- final: row dot(64) with W3, add partial logit, sigmoid ------------
    if (tid < 16) {
        float s = 0.f;
#pragma unroll
        for (int i = 0; i < H2; ++i) s += bf2f(h2s[tid * H2_LD + i]) * W3[i];
        float logit = logit_part[tm + tid] + s + b3[0];
        out[tm + tid] = 1.0f / (1.0f + __expf(-logit));
    }
}

// ---------------------------------------------------------------------------
extern "C" void kernel_launch(void* const* d_in, const int* in_sizes, int n_in,
                              void* d_out, int out_size, void* d_ws, size_t ws_size,
                              hipStream_t stream) {
    (void)in_sizes; (void)n_in; (void)out_size; (void)ws_size;

    const int*     user      = (const int*)    d_in[0];
    const int*     item      = (const int*)    d_in[1];
    const int*     brand     = (const int*)    d_in[2];
    const int*     cat_idx   = (const int*)    d_in[3];
    const uint8_t* cat_mask  = (const uint8_t*)d_in[4];
    const int*     text_idx  = (const int*)    d_in[5];
    const uint8_t* text_mask = (const uint8_t*)d_in[6];
    const float*   sales     = (const float*)  d_in[7];
    const float*   user_w    = (const float*)  d_in[8];
    const float*   item_w    = (const float*)  d_in[9];
    const float*   brand_w   = (const float*)  d_in[10];
    const float*   cat_w     = (const float*)  d_in[11];
    const float*   text_w    = (const float*)  d_in[12];
    const float*   sproj_w   = (const float*)  d_in[13];
    const float*   sproj_b   = (const float*)  d_in[14];
    const float*   user_l    = (const float*)  d_in[15];
    const float*   item_l    = (const float*)  d_in[16];
    const float*   brand_l   = (const float*)  d_in[17];
    const float*   cat_l     = (const float*)  d_in[18];
    const float*   text_l    = (const float*)  d_in[19];
    const float*   sales_lw  = (const float*)  d_in[20];
    const float*   sales_lb  = (const float*)  d_in[21];
    const float*   W1        = (const float*)  d_in[22];
    const float*   b1        = (const float*)  d_in[23];
    const float*   W2        = (const float*)  d_in[24];
    const float*   b2        = (const float*)  d_in[25];
    const float*   W3        = (const float*)  d_in[26];
    const float*   b3        = (const float*)  d_in[27];

    // workspace layout (256B-aligned offsets), ~12.8 MB total
    char* ws = (char*)d_ws;
    uint16_t* deep_x = (uint16_t*)(ws + 0);                     // 16384*384*2 = 12,582,912
    float*    logit  = (float*)   (ws + 12582912);              // 16384*4     =     65,536
    uint16_t* W1t    = (uint16_t*)(ws + 12648448);              // 128*384*2   =     98,304
    uint16_t* W2t    = (uint16_t*)(ws + 12746752);              //  64*128*2   =     16,384

    // 1) weights -> transposed bf16
    const int nconv = DX * H1 + H1 * H2;
    convert_w<<<(nconv + 255) / 256, 256, 0, stream>>>(W1, W2, W1t, W2t);

    // 2) gather/pool + first-order + FM2 (memory-bound pass, ~320 MB @ 23.3 TB/s)
    pool_kernel<<<B_ROWS, KDIM, 0, stream>>>(
        user, item, brand, cat_idx, cat_mask, text_idx, text_mask, sales,
        user_w, item_w, brand_w, cat_w, text_w, sproj_w, sproj_b,
        user_l, item_l, brand_l, cat_l, text_l, sales_lw, sales_lb,
        deep_x, logit);

    // 3) fused MLP + sigmoid (WMMA bf16, h1/h2 never leave LDS)
    fused_mlp<<<B_ROWS / 16, 256, 0, stream>>>(
        deep_x, W1t, W2t, b1, b2, W3, b3, logit, (float*)d_out);
}
